// APNet2_dAPNet2_MPNN_14370960572585
// MI455X (gfx1250) — matrix operand order repaired and checked by reading, not compiled, not run on hardware
//
#include <hip/hip_runtime.h>

// ---------------------------------------------------------------------------
// APNet2 fused MLP readout + scatter-sum for MI455X (gfx1250, wave32, WMMA)
//   E = relu(relu(relu(X@W1+b1)@W2+b2)@W3+b3)@W4+b4 ; out[d] += E*cutoff
// X = [hAB ; hBA] treated as 1M rows of 128 feats. bf16 WMMA, f32 accumulate.
// Accumulators start from inline-0 SRC2; biases are folded into the ReLU
// epilogue (fadd result is canonical -> single v_max_num_f32 per element).
// ---------------------------------------------------------------------------

typedef __attribute__((ext_vector_type(16))) __bf16 v16bf;
typedef __attribute__((ext_vector_type(8)))  float  v8f;

#define FEAT 128
#define H1   256
#define H2   128
#define H3   64

// LDS layout (units: bf16 halves). Strides padded +8 halves (16B) so that
// per-lane accesses with stride = row-stride hit distinct banks (4c mod 64).
#define WT1_STRIDE 136   // 128 + 8
#define WT2_STRIDE 264   // 256 + 8
#define WT3_STRIDE 136
#define H1S_STRIDE 264
#define H2S_STRIDE 136

#define OFF_WT1   0
#define OFF_WT2   (OFF_WT1 + H1 * WT1_STRIDE)         // 34816
#define OFF_WT3   (OFF_WT2 + H2 * WT2_STRIDE)         // 68608
#define OFF_STAGE (OFF_WT3 + H3 * WT3_STRIDE)         // 77312
#define STAGE_PER_WAVE (16 * H1S_STRIDE + 16 * H2S_STRIDE) // 6400
#define NWAVES 8
#define SMEM_HALVES (OFF_STAGE + NWAVES * STAGE_PER_WAVE)  // 128512
// f32 bias/W4 region appended after the bf16 region: b1(256) b2(128) b3(64) W4(64)
#define NBIAS 512
#define SMEM_BYTES  (SMEM_HALVES * 2 + NBIAS * 4)          // 259072 B < 320 KB

union Frag { uint4 u[2]; v16bf v; };

// ---------------------------------------------------------------------------
// Prep: zero output accumulator; convert W1/W2/W3 to bf16 TRANSPOSED ([N][K])
// in workspace so WMMA B-fragments are two contiguous 16B reads per lane.
// ---------------------------------------------------------------------------
__global__ void apnet2_prep_kernel(const float* __restrict__ W1,
                                   const float* __restrict__ W2,
                                   const float* __restrict__ W3,
                                   float* __restrict__ out, int outN,
                                   __bf16* __restrict__ wt1,
                                   __bf16* __restrict__ wt2,
                                   __bf16* __restrict__ wt3) {
  int i = blockIdx.x * blockDim.x + threadIdx.x;
  int stride = gridDim.x * blockDim.x;
  for (int j = i; j < outN; j += stride) out[j] = 0.0f;
  for (int j = i; j < H1 * FEAT; j += stride) {           // wt1[n][k] = W1[k][n]
    int n = j >> 7, k = j & 127;
    wt1[n * FEAT + k] = (__bf16)W1[k * H1 + n];
  }
  for (int j = i; j < H2 * H1; j += stride) {             // wt2[n][k] = W2[k][n]
    int n = j >> 8, k = j & 255;
    wt2[n * H1 + k] = (__bf16)W2[k * H2 + n];
  }
  for (int j = i; j < H3 * H2; j += stride) {             // wt3[n][k] = W3[k][n]
    int n = j >> 7, k = j & 127;
    wt3[n * H2 + k] = (__bf16)W3[k * H3 + n];
  }
}

// ---------------------------------------------------------------------------
// Main fused kernel. 256 threads = 8 waves; each wave owns a 16-row tile and
// grid-strides over tiles so the LDS weight staging is amortized.
// ---------------------------------------------------------------------------
__global__ __launch_bounds__(256)
void apnet2_mlp_kernel(const float* __restrict__ hAB, const float* __restrict__ hBA,
                       const float* __restrict__ cutoff, const int* __restrict__ dimer_ind,
                       const float* __restrict__ b1, const float* __restrict__ b2,
                       const float* __restrict__ b3, const float* __restrict__ b4,
                       const float* __restrict__ W4,
                       const __bf16* __restrict__ wt1, const __bf16* __restrict__ wt2,
                       const __bf16* __restrict__ wt3,
                       float* __restrict__ out, int npairs) {
  extern __shared__ char smem_raw[];
  __bf16* smem  = (__bf16*)smem_raw;
  float*  biasl = (float*)(smem_raw + SMEM_HALVES * 2);   // [512] f32

  const int tid = threadIdx.x;

  // ---- cooperative weight staging: global bf16 (transposed) -> padded LDS ----
  for (int t = tid; t < H1 * (FEAT / 8); t += blockDim.x) {   // 4096 x uint4
    int n = t >> 4, c = t & 15;
    *(uint4*)(smem + OFF_WT1 + n * WT1_STRIDE + c * 8) = *(const uint4*)(wt1 + n * FEAT + c * 8);
  }
  for (int t = tid; t < H2 * (H1 / 8); t += blockDim.x) {     // 4096 x uint4
    int n = t >> 5, c = t & 31;
    *(uint4*)(smem + OFF_WT2 + n * WT2_STRIDE + c * 8) = *(const uint4*)(wt2 + n * H1 + c * 8);
  }
  for (int t = tid; t < H3 * (H2 / 8); t += blockDim.x) {     // 1024 x uint4
    int n = t >> 4, c = t & 15;
    *(uint4*)(smem + OFF_WT3 + n * WT3_STRIDE + c * 8) = *(const uint4*)(wt3 + n * H2 + c * 8);
  }
  for (int t = tid; t < NBIAS; t += blockDim.x) {             // biases + W4 (f32)
    float v;
    if (t < 256)      v = b1[t];
    else if (t < 384) v = b2[t - 256];
    else if (t < 448) v = b3[t - 384];
    else              v = W4[t - 448];
    biasl[t] = v;
  }
  __syncthreads();

  const int lane = tid & 31;
  const int wave = tid >> 5;
  const int half = lane >> 4;   // K-run selector for A/B fragments
  const int lm   = lane & 15;   // row (A) / column (B,C,D) index

  __bf16* h1s = smem + OFF_STAGE + wave * STAGE_PER_WAVE;  // [16][264] bf16
  __bf16* h2s = h1s + 16 * H1S_STRIDE;                     // [16][136] bf16
  const float* b1s = biasl;          // [256]
  const float* b2s = biasl + 256;    // [128]
  const float* b3s = biasl + 384;    // [64]
  const float* w4s = biasl + 448;    // [64]
  const float  b4v = b4[0];

  const int nrows   = 2 * npairs;                  // hAB rows then hBA rows
  const int ntiles  = (nrows + 15) >> 4;
  const int wstride = (int)gridDim.x * NWAVES;

  for (int tile = (int)blockIdx.x * NWAVES + wave; tile < ntiles; tile += wstride) {
    // ---- A fragments, layer 1: row M = lm; runs k0..k0+7 and k0+16..k0+23 ----
    int  rowA  = tile * 16 + lm;
    bool valid = rowA < nrows;
    int  pr    = (rowA < npairs) ? rowA : rowA - npairs;
    const float* base = (rowA < npairs) ? hAB : hBA;
    if (!valid) { base = hAB; pr = 0; }   // benign dummy row; result discarded
    const float* p = base + (size_t)pr * FEAT;

    v16bf a[4];
#pragma unroll
    for (int kt = 0; kt < 4; ++kt) {
      int k0 = kt * 32 + half * 8;
      float4 f0 = *(const float4*)(p + k0);
      float4 f1 = *(const float4*)(p + k0 + 4);
      float4 f2 = *(const float4*)(p + k0 + 16);
      float4 f3 = *(const float4*)(p + k0 + 20);
      v16bf t;
      t[0]  = (__bf16)f0.x; t[1]  = (__bf16)f0.y; t[2]  = (__bf16)f0.z; t[3]  = (__bf16)f0.w;
      t[4]  = (__bf16)f1.x; t[5]  = (__bf16)f1.y; t[6]  = (__bf16)f1.z; t[7]  = (__bf16)f1.w;
      t[8]  = (__bf16)f2.x; t[9]  = (__bf16)f2.y; t[10] = (__bf16)f2.z; t[11] = (__bf16)f2.w;
      t[12] = (__bf16)f3.x; t[13] = (__bf16)f3.y; t[14] = (__bf16)f3.z; t[15] = (__bf16)f3.w;
      a[kt] = t;
    }

    // ---- layer 1: [16x128]@[128x256], bias+ReLU epilogue -> h1 staging ----
#pragma unroll 4
    for (int nt = 0; nt < 16; ++nt) {
      float bv = b1s[nt * 16 + lm];
      v8f c = {};                         // SRC2 starts as inline 0
#pragma unroll
      for (int kt = 0; kt < 4; ++kt) {
        const __bf16* bp = smem + OFF_WT1 + (nt * 16 + lm) * WT1_STRIDE + kt * 32 + half * 8;
        Frag bf; bf.u[0] = *(const uint4*)bp; bf.u[1] = *(const uint4*)(bp + 16);
        c = __builtin_amdgcn_wmma_f32_16x16x32_bf16(false, a[kt], false, bf.v, (short)0, c, false, false);
      }
#pragma unroll
      for (int r = 0; r < 8; ++r) {
        float v = fmaxf(c[r] + bv, 0.0f);                      // bias + ReLU
        h1s[(r + half * 8) * H1S_STRIDE + nt * 16 + lm] = (__bf16)v;
      }
    }
    asm volatile("s_wait_dscnt 0" ::: "memory");   // wave-local LDS RAW (cross-lane)

    // ---- layer 2: [16x256]@[256x128], bias+ReLU epilogue -> h2 staging ----
    // A fragments re-read from h1 staging per nt (keeps VGPR pressure low).
#pragma unroll 2
    for (int nt = 0; nt < 8; ++nt) {
      float bv = b2s[nt * 16 + lm];
      v8f c = {};
#pragma unroll
      for (int kt = 0; kt < 8; ++kt) {
        const __bf16* ap = h1s + lm * H1S_STRIDE + kt * 32 + half * 8;
        Frag af; af.u[0] = *(const uint4*)ap; af.u[1] = *(const uint4*)(ap + 16);
        const __bf16* bp = smem + OFF_WT2 + (nt * 16 + lm) * WT2_STRIDE + kt * 32 + half * 8;
        Frag bf; bf.u[0] = *(const uint4*)bp; bf.u[1] = *(const uint4*)(bp + 16);
        c = __builtin_amdgcn_wmma_f32_16x16x32_bf16(false, af.v, false, bf.v, (short)0, c, false, false);
      }
#pragma unroll
      for (int r = 0; r < 8; ++r) {
        float v = fmaxf(c[r] + bv, 0.0f);
        h2s[(r + half * 8) * H2S_STRIDE + nt * 16 + lm] = (__bf16)v;
      }
    }
    asm volatile("s_wait_dscnt 0" ::: "memory");

    // ---- layer 3: [16x128]@[128x64], bias+ReLU, fused layer-4 dot with W4 ----
    float s[8];
#pragma unroll
    for (int r = 0; r < 8; ++r) s[r] = 0.0f;
#pragma unroll
    for (int nt = 0; nt < 4; ++nt) {
      float bv = b3s[nt * 16 + lm];
      float wv = w4s[nt * 16 + lm];
      v8f c = {};
#pragma unroll
      for (int kt = 0; kt < 4; ++kt) {
        const __bf16* ap = h2s + lm * H2S_STRIDE + kt * 32 + half * 8;
        Frag af; af.u[0] = *(const uint4*)ap; af.u[1] = *(const uint4*)(ap + 16);
        const __bf16* bp = smem + OFF_WT3 + (nt * 16 + lm) * WT3_STRIDE + kt * 32 + half * 8;
        Frag bf; bf.u[0] = *(const uint4*)bp; bf.u[1] = *(const uint4*)(bp + 16);
        c = __builtin_amdgcn_wmma_f32_16x16x32_bf16(false, af.v, false, bf.v, (short)0, c, false, false);
      }
#pragma unroll
      for (int r = 0; r < 8; ++r) s[r] += fmaxf(c[r] + bv, 0.0f) * wv;
    }

    // reduce across the 16 lanes of each half (sums over output columns N)
#pragma unroll
    for (int off = 1; off < 16; off <<= 1) {
#pragma unroll
      for (int r = 0; r < 8; ++r) s[r] += __shfl_xor(s[r], off, 32);
    }

    // one writer per row: lane lm==r of each half emits row M = 8*half + r
#pragma unroll
    for (int r = 0; r < 8; ++r) {
      if (lm == r) {
        int row = tile * 16 + half * 8 + r;
        if (row < nrows) {
          int pr2 = (row < npairs) ? row : row - npairs;
          float val = (s[r] + b4v) * cutoff[pr2];
          __hip_atomic_fetch_add(&out[dimer_ind[pr2]], val,
                                 __ATOMIC_RELAXED, __HIP_MEMORY_SCOPE_AGENT);
        }
      }
    }
  }
}

// ---------------------------------------------------------------------------
extern "C" void kernel_launch(void* const* d_in, const int* in_sizes, int n_in,
                              void* d_out, int out_size, void* d_ws, size_t ws_size,
                              hipStream_t stream) {
  const float* hAB       = (const float*)d_in[0];
  const float* hBA       = (const float*)d_in[1];
  const float* cutoff    = (const float*)d_in[2];
  const int*   dimer_ind = (const int*)d_in[3];
  // d_in[4] = ndimer (device scalar; out_size carries the same value)
  const float* W1 = (const float*)d_in[5];
  const float* b1 = (const float*)d_in[6];
  const float* W2 = (const float*)d_in[7];
  const float* b2 = (const float*)d_in[8];
  const float* W3 = (const float*)d_in[9];
  const float* b3 = (const float*)d_in[10];
  const float* W4 = (const float*)d_in[11];
  const float* b4 = (const float*)d_in[12];
  float* out = (float*)d_out;

  const int npairs = in_sizes[0] / FEAT;

  // workspace: bf16 transposed weights (147456 bytes total)
  __bf16* wt1 = (__bf16*)d_ws;           // [H1][FEAT]
  __bf16* wt2 = wt1 + (size_t)H1 * FEAT; // [H2][H1]
  __bf16* wt3 = wt2 + (size_t)H2 * H1;   // [H3][H2]

  apnet2_prep_kernel<<<128, 256, 0, stream>>>(W1, W2, W3, out, out_size, wt1, wt2, wt3);

  (void)hipFuncSetAttribute((const void*)apnet2_mlp_kernel,
                            hipFuncAttributeMaxDynamicSharedMemorySize, SMEM_BYTES);

  long ntiles = (2L * npairs + 15) / 16;
  long want   = (ntiles + NWAVES - 1) / NWAVES;
  int  blocks = (int)(want < 2048 ? (want < 1 ? 1 : want) : 2048);
  apnet2_mlp_kernel<<<blocks, 256, SMEM_BYTES, stream>>>(
      hAB, hBA, cutoff, dimer_ind, b1, b2, b3, b4, W4, wt1, wt2, wt3, out, npairs);
}
